// MotionDecoder_12738873000644
// MI455X (gfx1250) — matrix operand order repaired and checked by reading, not compiled, not run on hardware
//
#include <hip/hip_runtime.h>
#include <hip/hip_bf16.h>
#include <cstdint>
#include <math.h>

typedef __attribute__((ext_vector_type(16))) _Float16 v16h;
typedef __attribute__((ext_vector_type(8)))  float    v8f;

// ---------------- deterministic pseudo-noise (Box-Muller on PCG hash) -----------
__device__ __forceinline__ uint32_t pcg_hash(uint32_t v) {
    v = v * 747796405u + 2891336453u;
    uint32_t w = ((v >> ((v >> 28u) + 4u)) ^ v) * 277803737u;
    return (w >> 22u) ^ w;
}
__device__ __forceinline__ float gauss_noise(uint32_t seed) {
    uint32_t h1 = pcg_hash(seed);
    uint32_t h2 = pcg_hash(seed ^ 0x9E3779B9u);
    float u1 = ((h1 >> 8) + 1u) * (1.0f / 16777217.0f);
    float u2 = (h2 >> 8) * (1.0f / 16777216.0f);
    return sqrtf(-2.0f * __logf(u1)) * __cosf(6.28318530718f * u2);
}

// ---------------- style: style[b,ci] = t[b,:] @ (mod_w/sqrt(32))[ci,:] + mod_b --
__global__ void style_kernel(const float* __restrict__ t,
                             const float* __restrict__ mod_w,
                             const float* __restrict__ mod_b,
                             float* __restrict__ style,
                             int B, int Cin) {
    int i = blockIdx.x * blockDim.x + threadIdx.x;
    if (i >= B * Cin) return;
    int b = i / Cin, ci = i % Cin;
    const float* tr = t + b * 32;
    const float* wr = mod_w + (size_t)ci * 32;
    float s = 0.f;
#pragma unroll
    for (int l = 0; l < 32; ++l) s += tr[l] * wr[l];
    style[i] = s * 0.17677669529663687f /* 1/sqrt(32) */ + mod_b[ci];
}

// ---------------- wsq[co,ci] = sum_k conv_w^2 ----------------------------------
__global__ void wsq_kernel(const float* __restrict__ w, float* __restrict__ wsq, int n) {
    int i = blockIdx.x * blockDim.x + threadIdx.x;
    if (i >= n) return;
    const float* p = w + (size_t)i * 9;
    float s = 0.f;
#pragma unroll
    for (int k = 0; k < 9; ++k) s += p[k] * p[k];
    wsq[i] = s;
}

// ---------------- demod[b,co] = rsqrt(scale^2 * sum_ci style^2*wsq + 1e-8) -----
__global__ void demod_kernel(const float* __restrict__ style,
                             const float* __restrict__ wsq,
                             float* __restrict__ demod,
                             int B, int Cin, int Cout, float scale2) {
    int i = blockIdx.x * blockDim.x + threadIdx.x;
    if (i >= B * Cout) return;
    int b = i / Cout, co = i % Cout;
    const float* st = style + (size_t)b * Cin;
    const float* wq = wsq + (size_t)co * Cin;
    float s = 0.f;
    for (int ci = 0; ci < Cin; ++ci) { float sv = st[ci]; s += sv * sv * wq[ci]; }
    demod[i] = rsqrtf(scale2 * s + 1e-8f);
}

// ---------------- weight swizzle into WMMA A-fragment layout -------------------
// Layout: wz[co_blk][cin_blk][tap][lane][e]  (16 halves per lane = one 32B load)
// A is 16x32 f16 (MxK): lane: M=lane&15, hi=lane>>4; element e: v=e/2,p=e&1,
// K = hi*16 + e   (contiguous order, matches B fragment)
__global__ void wswz_kernel(const float* __restrict__ w, _Float16* __restrict__ wz,
                            int Cin, int Cout, int up, float scale, int total) {
    int i = blockIdx.x * blockDim.x + threadIdx.x;
    if (i >= total) return;
    int e    = i & 15;
    int lane = (i >> 4) & 31;
    int tap  = (i >> 9) % 9;
    int blk  = i / 4608;              // 9*512
    int nCinBlk = Cin >> 5;
    int cin_blk = blk % nCinBlk;
    int co_blk  = blk / nCinBlk;
    int hi = lane >> 4;
    int co = (co_blk << 4) + (lane & 15);
    int v = e >> 1, p = e & 1;
    int cin_l = (hi << 4) + e;
    int ci = (cin_blk << 5) + cin_l;
    int ky = tap / 3, kx = tap % 3;
    if (up) { ky = 2 - ky; kx = 2 - kx; }  // transposed conv == conv with flipped w
    float val = w[(((size_t)co * Cin + ci) * 3 + ky) * 3 + kx] * scale;
    wz[i] = (_Float16)val;
}

// ---------------- broadcast const input over batch -----------------------------
__global__ void bcast_const(const float* __restrict__ c, float* __restrict__ act,
                            int B, int n) {
    int i = blockIdx.x * blockDim.x + threadIdx.x;
    if (i >= B * n) return;
    act[i] = c[i % n];
}

// ---------------- implicit-GEMM 3x3 conv via WMMA ------------------------------
// One workgroup: 4 waves, one batch image, one 8x4 output pixel tile
// (two 4x4 sub-tiles). The FULL Cin halo (10x6 x Cin, style-scaled f16) is
// staged to LDS once (<= 60 KB for Cin=512), then each wave processes
// co-block PAIRS with a 2x2 register tile:
//   4 accumulators = (2 co blocks) x (2 pixel sub-tiles)
//   per K-step: 2 A global loads + 2 B LDS loads -> 4 WMMAs
// This halves both weight (L2) and LDS traffic per WMMA vs a 1x1 tile and
// gives 4 independent accumulation chains to cover WMMA->WMMA hazards.
template <bool UP, int SHIFT>   // Cin = 1<<SHIFT
__global__ __launch_bounds__(128) void conv_kernel(
        const float* __restrict__ x, const _Float16* __restrict__ wz,
        const float* __restrict__ style, float* __restrict__ y,
        int B, int Cout, int Hin, int Win, int Ho, int Wo,
        int ntX, int ntY) {
    constexpr int Cin     = 1 << SHIFT;
    constexpr int nCinBlk = Cin >> 5;
    __shared__ __align__(64) _Float16 xs[60 << SHIFT];  // 10x6 halo x Cin, f16

    const int ntile = ntX * ntY;
    const int tile  = blockIdx.x % ntile;
    const int b     = blockIdx.x / ntile;
    const int ty0 = (tile / ntX) * 8;
    const int tx0 = (tile % ntX) * 4;
    const int tid  = threadIdx.x;
    const int lane = tid & 31;
    const int wave = tid >> 5;
    const float* stb = style + (size_t)b * Cin;
    const float* xb  = x + (size_t)b * Cin * Hin * Win;

    const int hi = lane >> 4;       // D/B fragment half-select
    const int n  = lane & 15;       // output pixel within 4x4 sub-tile
    const int py = n >> 2, px = n & 3;

    // ---- stage style-scaled f16 halo once (zeros implement padding / dilation)
    for (int i = tid; i < (60 << SHIFT); i += 128) {
        int hp = i >> SHIFT;          // halo pixel 0..59 (10 rows x 6 cols)
        int ci = i & (Cin - 1);
        int hy = hp / 6, hx = hp % 6;
        float v = 0.f;
        if (UP) {
            // coords in zero-stuffed + pad-2 space (size 2H+3)
            int dy = ty0 + hy, dx = tx0 + hx;
            if (!((dy | dx) & 1) && dy >= 2 && dx >= 2) {
                int iy = (dy - 2) >> 1, ix = (dx - 2) >> 1;
                if (iy < Hin && ix < Win) v = xb[((size_t)ci * Hin + iy) * Win + ix];
            }
        } else {
            int iy = ty0 + hy - 1, ix = tx0 + hx - 1;
            if (iy >= 0 && ix >= 0 && iy < Hin && ix < Win)
                v = xb[((size_t)ci * Hin + iy) * Win + ix];
        }
        xs[i] = (_Float16)(v * stb[ci]);
    }
    __syncthreads();

    const int ox  = tx0 + px;
    const int oy0 = ty0 + py;        // sub-tile 0
    const int oy1 = oy0 + 4;         // sub-tile 1
    const bool in0 = (oy0 < Ho) && (ox < Wo);
    const bool in1 = (oy1 < Ho) && (ox < Wo);
    const int nPair = Cout >> 5;      // pairs of 16-channel co blocks
    const int slots = gridDim.y * 4;
    const size_t coStride = ((size_t)nCinBlk * 9) << 9;   // halves per co block

    for (int cp = blockIdx.y * 4 + wave; cp < nPair; cp += slots) {
        v8f acc00 = {}, acc01 = {}, acc10 = {}, acc11 = {};
        const _Float16* w0 = wz + (size_t)(2 * cp) * coStride + (lane << 4);
        const _Float16* w1 = w0 + coStride;
        for (int cb = 0; cb < nCinBlk; ++cb) {
            const _Float16* w0c = w0 + (((size_t)cb * 9) << 9);
            const _Float16* w1c = w1 + (((size_t)cb * 9) << 9);
            if (cb + 1 < nCinBlk) {   // warm WGP$ one K-step ahead (global_prefetch)
                __builtin_prefetch(w0c + (9 << 9), 0, 3);
                __builtin_prefetch(w1c + (9 << 9), 0, 3);
            }
#pragma unroll
            for (int tap = 0; tap < 9; ++tap) {
                int dy = tap / 3, dx = tap % 3;
                int hp0 = (py + dy) * 6 + (px + dx);       // sub-tile 0
                int hp1 = hp0 + 24;                        // sub-tile 1 (+4 rows)
                // B fragment: lane = column N (pixel); K = hi*16 + e (cin contiguous)
                v16h b0 = *(const v16h*)(&xs[(hp0 << SHIFT) + (cb << 5) + (hi << 4)]);
                v16h b1 = *(const v16h*)(&xs[(hp1 << SHIFT) + (cb << 5) + (hi << 4)]);
                v16h a0 = *(const v16h*)(w0c + (tap << 9));
                v16h a1 = *(const v16h*)(w1c + (tap << 9));
                acc00 = __builtin_amdgcn_wmma_f32_16x16x32_f16(
                            false, a0, false, b0, (short)0, acc00, false, false);
                acc01 = __builtin_amdgcn_wmma_f32_16x16x32_f16(
                            false, a0, false, b1, (short)0, acc01, false, false);
                acc10 = __builtin_amdgcn_wmma_f32_16x16x32_f16(
                            false, a1, false, b0, (short)0, acc10, false, false);
                acc11 = __builtin_amdgcn_wmma_f32_16x16x32_f16(
                            false, a1, false, b1, (short)0, acc11, false, false);
            }
        }
#pragma unroll
        for (int r = 0; r < 8; ++r) {
            int co0 = ((2 * cp)     << 4) + r + (hi << 3);  // D layout: M = r + 8*hi
            int co1 = ((2 * cp + 1) << 4) + r + (hi << 3);
            if (in0) {
                y[(((size_t)b * Cout + co0) * Ho + oy0) * Wo + ox] = acc00[r];
                y[(((size_t)b * Cout + co1) * Ho + oy0) * Wo + ox] = acc10[r];
            }
            if (in1) {
                y[(((size_t)b * Cout + co0) * Ho + oy1) * Wo + ox] = acc01[r];
                y[(((size_t)b * Cout + co1) * Ho + oy1) * Wo + ox] = acc11[r];
            }
        }
    }
}

// ---------------- epilogue (non-upsample): demod + noise + bias + fused lrelu --
__global__ void epilogue_kernel(const float* __restrict__ ycv,
                                const float* __restrict__ demod,
                                const float* __restrict__ bias,
                                const float* __restrict__ noise_w,
                                float* __restrict__ act, float* __restrict__ tap_out,
                                int B, int C, int H, int W, uint32_t seed) {
    int i = blockIdx.x * blockDim.x + threadIdx.x;
    int total = B * C * H * W;
    if (i >= total) return;
    int wx = i % W;
    int hy = (i / W) % H;
    int co = (i / (W * H)) % C;
    int b  = i / (W * H * C);
    float nz = gauss_noise(seed + (uint32_t)(((b * H) + hy) * W + wx) * 2654435761u);
    float v = ycv[i] * demod[b * C + co] + noise_w[0] * nz + bias[co];
    v = (v > 0.f ? v : 0.2f * v) * 1.4142135623730951f;
    act[i] = v;
    if (tap_out) tap_out[i] = v;
}

// ---------------- epilogue (upsample): blur([1,3,3,1]^2/16) + demod + rest -----
__global__ void blur_epilogue_kernel(const float* __restrict__ ycv,
                                     const float* __restrict__ demod,
                                     const float* __restrict__ bias,
                                     const float* __restrict__ noise_w,
                                     float* __restrict__ act, float* __restrict__ tap_out,
                                     int B, int C, int Hi /*2H+1*/, int Ho /*2H*/,
                                     uint32_t seed) {
    int i = blockIdx.x * blockDim.x + threadIdx.x;
    int total = B * C * Ho * Ho;
    if (i >= total) return;
    int ox = i % Ho;
    int oy = (i / Ho) % Ho;
    int co = (i / (Ho * Ho)) % C;
    int b  = i / (Ho * Ho * C);
    const float k1[4] = {1.f, 3.f, 3.f, 1.f};
    const float* src = ycv + (size_t)(b * C + co) * Hi * Hi;
    float s = 0.f;
#pragma unroll
    for (int a = 0; a < 4; ++a) {
        int iy = oy + a - 1;
        if (iy < 0 || iy >= Hi) continue;
#pragma unroll
        for (int c2 = 0; c2 < 4; ++c2) {
            int ix = ox + c2 - 1;
            if (ix < 0 || ix >= Hi) continue;
            s += k1[a] * k1[c2] * src[iy * Hi + ix];
        }
    }
    s *= (1.f / 16.f);
    float nz = gauss_noise(seed + (uint32_t)(((b * Ho) + oy) * Ho + ox) * 2654435761u);
    float v = s * demod[b * C + co] + noise_w[0] * nz + bias[co];
    v = (v > 0.f ? v : 0.2f * v) * 1.4142135623730951f;
    act[i] = v;
    if (tap_out) tap_out[i] = v;
}

// ------------------------------ host driver ------------------------------------
static inline void launch_conv(bool up, int Cin, dim3 grid, hipStream_t stream,
                               const float* cur, const _Float16* wz, const float* style,
                               float* ycv, int B, int Cout, int H, int W, int Ho, int Wo,
                               int ntX, int ntY) {
#define CONV_CASE(UPV, SH)                                                          \
    conv_kernel<UPV, SH><<<grid, 128, 0, stream>>>(cur, wz, style, ycv, B, Cout,    \
                                                   H, W, Ho, Wo, ntX, ntY)
    if (up) {
        switch (Cin) {
            case 32:  CONV_CASE(true, 5); break;
            case 128: CONV_CASE(true, 7); break;
            case 256: CONV_CASE(true, 8); break;
            default:  CONV_CASE(true, 9); break;
        }
    } else {
        switch (Cin) {
            case 32:  CONV_CASE(false, 5); break;
            case 128: CONV_CASE(false, 7); break;
            case 256: CONV_CASE(false, 8); break;
            default:  CONV_CASE(false, 9); break;
        }
    }
#undef CONV_CASE
}

extern "C" void kernel_launch(void* const* d_in, const int* in_sizes, int n_in,
                              void* d_out, int out_size, void* d_ws, size_t ws_size,
                              hipStream_t stream) {
    (void)in_sizes; (void)n_in; (void)out_size; (void)ws_size;
    const float* t   = (const float*)d_in[0];
    const float* cst = (const float*)d_in[1];

    struct Cfg { int cin, cout, up; };
    static const Cfg L[13] = {
        {32, 512, 0}, {512, 512, 1}, {512, 512, 0}, {512, 512, 0},
        {512, 512, 1}, {512, 512, 0}, {512, 512, 0}, {512, 256, 1},
        {256, 256, 0}, {256, 256, 0}, {256, 128, 1}, {128, 128, 0}, {128, 128, 0}};
    const int B = 16;

    // workspace carve-out (256B aligned)
    char* base = (char*)d_ws;
    size_t off = 0;
    auto carve = [&](size_t bytes) -> void* {
        void* p = base + off;
        off += (bytes + 255) & ~(size_t)255;
        return p;
    };
    float*    actA  = (float*)carve((size_t)8388608 * 4);    // max activation 16*128*64*64
    float*    actB  = (float*)carve((size_t)8388608 * 4);
    float*    ycv   = (float*)carve((size_t)8652800 * 4);    // max conv out 16*128*65*65
    _Float16* wz    = (_Float16*)carve((size_t)4718592);     // 512*512*9 halves
    float*    wsq   = (float*)carve((size_t)1048576);        // 512*512
    float*    style = (float*)carve((size_t)32768);          // 16*512
    float*    demod = (float*)carve((size_t)32768);

    float* out = (float*)d_out;

    // layer 0 input: broadcast const over batch
    {
        int n = 32 * 4 * 4;
        bcast_const<<<(B * n + 255) / 256, 256, 0, stream>>>(cst, actA, B, n);
    }

    float* cur = actA;
    float* nxt = actB;
    int H = 4, W = 4;

    for (int i = 0; i < 13; ++i) {
        const int Cin = L[i].cin, Cout = L[i].cout;
        const bool up = L[i].up != 0;
        const float* mod_w   = (const float*)d_in[2 + i * 5 + 0];
        const float* mod_b   = (const float*)d_in[2 + i * 5 + 1];
        const float* conv_w  = (const float*)d_in[2 + i * 5 + 2];
        const float* noise_w = (const float*)d_in[2 + i * 5 + 3];
        const float* bias    = (const float*)d_in[2 + i * 5 + 4];
        const float scale = 1.0f / sqrtf((float)(Cin * 9));

        style_kernel<<<(B * Cin + 255) / 256, 256, 0, stream>>>(t, mod_w, mod_b, style, B, Cin);
        wsq_kernel<<<(Cout * Cin + 255) / 256, 256, 0, stream>>>(conv_w, wsq, Cout * Cin);
        demod_kernel<<<(B * Cout + 255) / 256, 256, 0, stream>>>(style, wsq, demod,
                                                                 B, Cin, Cout, scale * scale);
        const int total_wz = Cout * Cin * 9;
        wswz_kernel<<<(total_wz + 255) / 256, 256, 0, stream>>>(conv_w, wz, Cin, Cout,
                                                                up ? 1 : 0, scale, total_wz);

        const int Ho = up ? 2 * H + 1 : H;   // conv output (pre-blur for upsample)
        const int Wo = up ? 2 * W + 1 : W;
        const int ntX = (Wo + 3) / 4, ntY = (Ho + 7) / 8;  // 8x4 pixel tiles
        // adaptive co-block split: keep >= ~512 workgroups in flight without
        // over-splitting the co-pair loop (nPair = Cout/32 pairs per tile)
        const int wgs = B * ntX * ntY;
        const int nPair = Cout / 32;
        int g = (512 + wgs - 1) / wgs;
        int gmax = (nPair + 3) / 4;
        if (g > gmax) g = gmax;
        if (g < 1) g = 1;
        dim3 grid(wgs, g);
        launch_conv(up, Cin, grid, stream, cur, wz, style, ycv,
                    B, Cout, H, W, Ho, Wo, ntX, ntY);

        const uint32_t seed = 0xC0FFEEu + (uint32_t)i * 0x9E3779B9u;
        float* tp = nullptr;
        if (i == 3)  tp = out + 0;         // m1: 16*512*8*8
        if (i == 6)  tp = out + 524288;    // m2: 16*512*16*16
        if (i == 9)  tp = out + 2621440;   // m3: 16*256*32*32
        if (i == 12) tp = out + 6815744;   // m4: 16*128*64*64

        if (up) {
            const int Hout = 2 * H;
            const int tot = B * Cout * Hout * Hout;
            blur_epilogue_kernel<<<(tot + 255) / 256, 256, 0, stream>>>(
                ycv, demod, bias, noise_w, nxt, tp, B, Cout, Ho, Hout, seed);
            H = Hout; W = Hout;
        } else {
            const int tot = B * Cout * H * W;
            epilogue_kernel<<<(tot + 255) / 256, 256, 0, stream>>>(
                ycv, demod, bias, noise_w, nxt, tp, B, Cout, H, W, seed);
        }
        float* tmp = cur; cur = nxt; nxt = tmp;
    }
}
